// SingleHeadAttention_17901423690239
// MI455X (gfx1250) — matrix operand order repaired and checked by reading, not compiled, not run on hardware
//
#include <hip/hip_runtime.h>
#include <hip/hip_bf16.h>
#include <math.h>

// ---------------------------------------------------------------------------
// Single-head attention, MI455X (gfx1250, wave32, WMMA bf16 f32-accumulate).
//   B=4, T=2048, D_MODEL=D_K=D_V=1024.
// v2: register-blocked GEMMs to cut the vmem:wmma issue ratio
//   - projection: 2x4 tile block per wave (8 WMMA per 12 b128 loads)
//   - attention scores: 4 s-tiles share one Q A-frag (4 WMMA per 10 loads)
//   - attention PV: 8 n-tiles share one P A-frag (8 WMMA per 18 loads)
// ---------------------------------------------------------------------------

typedef __bf16 bf16_t;
typedef __attribute__((ext_vector_type(16))) __bf16 v16bf;
typedef __attribute__((ext_vector_type(8)))  __bf16 v8bf;
typedef __attribute__((ext_vector_type(8)))  float  v8f;

#define BATCH   4
#define SEQ     2048
#define DMODEL  1024
#define DK      1024
#define DV      1024
#define BT      (BATCH * SEQ)          // 8192 rows
#define SCALE   0.03125f               // 1/sqrt(1024)

// ---------------------------------------------------------------------------
// Fragment helpers (wave32 lane layouts used by this code; A/B share contiguous k order)
// A (16x32 bf16):  lane = M row; lanes 0-15 hold K = 0..15 contiguous,
//                  lanes 16-31 hold K = 16..31 contiguous (as v16bf elems 0..15).
// B (32x16 bf16):  lane = N col; lanes 0-15 hold K = 0..15 contiguous,
//                  lanes 16-31 hold K = 16..31 contiguous.
// C/D (16x16 f32): VGPR j: M = j + 8*(lane>>4), N = lane&15.
// ---------------------------------------------------------------------------
__device__ __forceinline__ v16bf make_a_frag(const bf16_t* row /* &A[m][k0] */,
                                             int kb /* (lane>>4)*8 */) {
    v8bf lo = *(const v8bf*)(row + 2 * kb);
    v8bf hi = *(const v8bf*)(row + 2 * kb + 8);
    v16bf r;
#pragma unroll
    for (int i = 0; i < 8; ++i) { r[i] = lo[i]; r[i + 8] = hi[i]; }
    return r;
}

__device__ __forceinline__ v16bf make_b_frag(const bf16_t* p /* 16 contig K */) {
    return *(const v16bf*)p;
}

__device__ __forceinline__ v8f wmma_bf16(v16bf a, v16bf b, v8f c) {
    return __builtin_amdgcn_wmma_f32_16x16x32_bf16(
        /*neg_a=*/false, a, /*neg_b=*/false, b,
        /*c_mod=*/(short)0, c, /*reuse_a=*/false, /*reuse_b=*/false);
}

// ---------------------------------------------------------------------------
// Kernel 1: f32 -> bf16 conversion
// ---------------------------------------------------------------------------
__global__ void cvt_f32_bf16_kernel(const float* __restrict__ in,
                                    bf16_t* __restrict__ out, int n) {
    int i = blockIdx.x * blockDim.x + threadIdx.x;
    if (i < n) out[i] = (bf16_t)in[i];
}

// ---------------------------------------------------------------------------
// Kernel 2: projection GEMM  Y = X * W^T   (register-blocked 32x64 per wave)
//   X: [BT, DMODEL] bf16 ; W: [1024, DMODEL] bf16 (row-major, as nn.Linear)
//   transposeStore == 0 : Y[r][c]                     ([BT, 1024])
//   transposeStore == 1 : Y[b][c][t]  (per-batch V^T) ([BATCH, 1024, SEQ])
// ---------------------------------------------------------------------------
__global__ void proj_gemm_kernel(const bf16_t* __restrict__ X,
                                 const bf16_t* __restrict__ W,
                                 bf16_t* __restrict__ Y,
                                 int transposeStore) {
    const int wave = blockIdx.x * (blockDim.x >> 5) + (threadIdx.x >> 5);
    const int rg   = wave >> 4;          // row group:  32 rows   (BT/32 = 256)
    const int cg   = wave & 15;          // col group:  64 cols   (1024/64 = 16)
    const int row0 = rg * 32;
    const int col0 = cg * 64;

    const int lane = threadIdx.x & 31;
    const int half = lane >> 4;
    const int lm   = lane & 15;
    const int kbA  = half * 8;
    const int kbB  = half * 16;

    const bf16_t* aRow0 = X + (size_t)(row0 + lm) * DMODEL;
    const bf16_t* aRow1 = aRow0 + (size_t)16 * DMODEL;
    const bf16_t* bRow0 = W + (size_t)(col0 + 0  + lm) * DMODEL;
    const bf16_t* bRow1 = W + (size_t)(col0 + 16 + lm) * DMODEL;
    const bf16_t* bRow2 = W + (size_t)(col0 + 32 + lm) * DMODEL;
    const bf16_t* bRow3 = W + (size_t)(col0 + 48 + lm) * DMODEL;

    v8f acc00 = {}, acc01 = {}, acc02 = {}, acc03 = {};
    v8f acc10 = {}, acc11 = {}, acc12 = {}, acc13 = {};

    for (int k0 = 0; k0 < DMODEL; k0 += 32) {
        v16bf a0 = make_a_frag(aRow0 + k0, kbA);
        v16bf a1 = make_a_frag(aRow1 + k0, kbA);
        v16bf b0 = make_b_frag(bRow0 + k0 + kbB);
        v16bf b1 = make_b_frag(bRow1 + k0 + kbB);
        v16bf b2 = make_b_frag(bRow2 + k0 + kbB);
        v16bf b3 = make_b_frag(bRow3 + k0 + kbB);
        acc00 = wmma_bf16(a0, b0, acc00);
        acc01 = wmma_bf16(a0, b1, acc01);
        acc02 = wmma_bf16(a0, b2, acc02);
        acc03 = wmma_bf16(a0, b3, acc03);
        acc10 = wmma_bf16(a1, b0, acc10);
        acc11 = wmma_bf16(a1, b1, acc11);
        acc12 = wmma_bf16(a1, b2, acc12);
        acc13 = wmma_bf16(a1, b3, acc13);
    }

    v8f accs[2][4] = {{acc00, acc01, acc02, acc03},
                      {acc10, acc11, acc12, acc13}};

    if (transposeStore == 0) {
#pragma unroll
        for (int mi = 0; mi < 2; ++mi) {
#pragma unroll
            for (int ci = 0; ci < 4; ++ci) {
#pragma unroll
                for (int j = 0; j < 8; ++j) {
                    const int M = j + half * 8;
                    Y[(size_t)(row0 + mi * 16 + M) * 1024 + col0 + ci * 16 + lm] =
                        (bf16_t)accs[mi][ci][j];
                }
            }
        }
    } else {
        const int bIdx = row0 / SEQ;     // 32-row block stays inside one batch
        const int t0   = row0 % SEQ;
#pragma unroll
        for (int mi = 0; mi < 2; ++mi) {
#pragma unroll
            for (int ci = 0; ci < 4; ++ci) {
#pragma unroll
                for (int j = 0; j < 8; ++j) {
                    const int M = j + half * 8;
                    Y[((size_t)bIdx * DV + col0 + ci * 16 + lm) * SEQ + t0 + mi * 16 + M] =
                        (bf16_t)accs[mi][ci][j];
                }
            }
        }
    }
}

// ---------------------------------------------------------------------------
// Kernel 3: fused attention for one (batch, 16-query-row) block.
//   grid = (SEQ/16, BATCH), block = 256 threads (8 waves).
//   LDS: S[16][2048] f32 (128KB) + P[16][2048] bf16 (64KB) + reduction scratch.
// ---------------------------------------------------------------------------
__global__ void attn_kernel(const bf16_t* __restrict__ Q,
                            const bf16_t* __restrict__ K,
                            const bf16_t* __restrict__ Vt,
                            float* __restrict__ O) {
    extern __shared__ char smem[];
    float*  S      = (float*)smem;                                   // 16*2048 f32
    bf16_t* P      = (bf16_t*)(smem + 16 * 2048 * 4);                // 16*2048 bf16
    float*  red    = (float*)(smem + 16 * 2048 * 4 + 16 * 2048 * 2); // 256 f32
    float*  rowmax = red + 256;                                      // 16 f32
    float*  rowsum = rowmax + 16;                                    // 16 f32

    const int b    = blockIdx.y;
    const int qt   = blockIdx.x;
    const int q0   = qt * 16;
    const int tid  = threadIdx.x;
    const int wave = tid >> 5;
    const int lane = tid & 31;
    const int half = lane >> 4;
    const int lm   = lane & 15;
    const int kbA  = half * 8;
    const int kbB  = half * 16;

    // ---- Phase 1: S = (Q Kt) * scale. Waves split 32 groups of 4 s-tiles. --
    // One Q A-frag feeds 4 WMMAs (Q frags identical across s-tiles).
    const bf16_t* qRow = Q + ((size_t)b * SEQ + q0 + lm) * DK;
    for (int sg = wave; sg < SEQ / 64; sg += 8) {
        const int s0c = sg * 64;
        const bf16_t* kRow0 = K + ((size_t)b * SEQ + s0c + 0  + lm) * DK;
        const bf16_t* kRow1 = K + ((size_t)b * SEQ + s0c + 16 + lm) * DK;
        const bf16_t* kRow2 = K + ((size_t)b * SEQ + s0c + 32 + lm) * DK;
        const bf16_t* kRow3 = K + ((size_t)b * SEQ + s0c + 48 + lm) * DK;

        v8f acc0 = {}, acc1 = {}, acc2 = {}, acc3 = {};
        for (int k0 = 0; k0 < DK; k0 += 32) {
            v16bf a = make_a_frag(qRow + k0, kbA);
            acc0 = wmma_bf16(a, make_b_frag(kRow0 + k0 + kbB), acc0);
            acc1 = wmma_bf16(a, make_b_frag(kRow1 + k0 + kbB), acc1);
            acc2 = wmma_bf16(a, make_b_frag(kRow2 + k0 + kbB), acc2);
            acc3 = wmma_bf16(a, make_b_frag(kRow3 + k0 + kbB), acc3);
        }

        v8f accs[4] = {acc0, acc1, acc2, acc3};
#pragma unroll
        for (int ci = 0; ci < 4; ++ci) {
#pragma unroll
            for (int j = 0; j < 8; ++j) {
                const int M = j + half * 8;
                S[M * SEQ + s0c + ci * 16 + lm] = accs[ci][j] * SCALE;
            }
        }
    }
    __syncthreads();

    // ---- Phase 2: row softmax over S (16 rows x 16 segments x 128 cols) ----
    {
        const int r   = tid >> 4;
        const int seg = tid & 15;
        float* Srow = S + r * SEQ + seg * 128;

        float m = -3.402823466e38f;
        for (int i = 0; i < 128; ++i) m = fmaxf(m, Srow[i]);
        red[tid] = m;
        __syncthreads();
        if (seg == 0) {
            float mm = red[r * 16];
            for (int i = 1; i < 16; ++i) mm = fmaxf(mm, red[r * 16 + i]);
            rowmax[r] = mm;
        }
        __syncthreads();

        const float rm = rowmax[r];
        float sum = 0.0f;
        for (int i = 0; i < 128; ++i) {
            float e = __expf(Srow[i] - rm);
            Srow[i] = e;
            sum += e;
        }
        red[tid] = sum;
        __syncthreads();
        if (seg == 0) {
            float ss = 0.0f;
            for (int i = 0; i < 16; ++i) ss += red[r * 16 + i];
            rowsum[r] = ss;
        }
        __syncthreads();

        const float inv = 1.0f / rowsum[r];
        bf16_t* Prow = P + r * SEQ + seg * 128;
        for (int i = 0; i < 128; ++i) Prow[i] = (bf16_t)(Srow[i] * inv);
    }
    __syncthreads();

    // ---- Phase 3: O = P * V. Each wave owns a 16x128 slab of O. ------------
    // A-frag from LDS (ds_load_b128 path); B-frag from global Vt rows.
    const bf16_t* pRow  = P + lm * SEQ;                 // A row for this lane
    const int     n0    = wave * 128;
    const bf16_t* vBase = Vt + ((size_t)b * DV + n0 + lm) * SEQ;

    v8f acc0 = {}, acc1 = {}, acc2 = {}, acc3 = {};
    v8f acc4 = {}, acc5 = {}, acc6 = {}, acc7 = {};

    for (int s0 = 0; s0 < SEQ; s0 += 32) {
        v16bf a = make_a_frag(pRow + s0, kbA);
        const bf16_t* vp = vBase + s0 + kbB;
        acc0 = wmma_bf16(a, make_b_frag(vp + (size_t)0 * 16 * SEQ), acc0);
        acc1 = wmma_bf16(a, make_b_frag(vp + (size_t)1 * 16 * SEQ), acc1);
        acc2 = wmma_bf16(a, make_b_frag(vp + (size_t)2 * 16 * SEQ), acc2);
        acc3 = wmma_bf16(a, make_b_frag(vp + (size_t)3 * 16 * SEQ), acc3);
        acc4 = wmma_bf16(a, make_b_frag(vp + (size_t)4 * 16 * SEQ), acc4);
        acc5 = wmma_bf16(a, make_b_frag(vp + (size_t)5 * 16 * SEQ), acc5);
        acc6 = wmma_bf16(a, make_b_frag(vp + (size_t)6 * 16 * SEQ), acc6);
        acc7 = wmma_bf16(a, make_b_frag(vp + (size_t)7 * 16 * SEQ), acc7);
    }

    float* oBase = O + ((size_t)b * SEQ + q0) * DV;
    v8f accs[8] = {acc0, acc1, acc2, acc3, acc4, acc5, acc6, acc7};
#pragma unroll
    for (int i = 0; i < 8; ++i) {
#pragma unroll
        for (int j = 0; j < 8; ++j) {
            const int M = j + half * 8;
            oBase[(size_t)M * DV + n0 + i * 16 + lm] = accs[i][j];
        }
    }
}

// ---------------------------------------------------------------------------
// Host-side launch. Workspace layout (bytes, ~73.4 MB total):
//   xb  [BT*DMODEL]  bf16  16.78 MB
//   Wqb/Wkb/Wvb      bf16   2.10 MB each
//   Qb  [BT*DK]      bf16  16.78 MB
//   Kb  [BT*DK]      bf16  16.78 MB
//   Vtb [BATCH*DV*SEQ] bf16 16.78 MB (per-batch transposed V)
// ---------------------------------------------------------------------------
extern "C" void kernel_launch(void* const* d_in, const int* in_sizes, int n_in,
                              void* d_out, int out_size, void* d_ws, size_t ws_size,
                              hipStream_t stream) {
    const float* x  = (const float*)d_in[0];
    const float* Wq = (const float*)d_in[1];
    const float* Wk = (const float*)d_in[2];
    const float* Wv = (const float*)d_in[3];
    float* out = (float*)d_out;

    char* ws = (char*)d_ws;
    size_t off = 0;
    bf16_t* xb  = (bf16_t*)(ws + off); off += (size_t)BT * DMODEL * 2;
    bf16_t* Wqb = (bf16_t*)(ws + off); off += (size_t)DK * DMODEL * 2;
    bf16_t* Wkb = (bf16_t*)(ws + off); off += (size_t)DK * DMODEL * 2;
    bf16_t* Wvb = (bf16_t*)(ws + off); off += (size_t)DV * DMODEL * 2;
    bf16_t* Qb  = (bf16_t*)(ws + off); off += (size_t)BT * DK * 2;
    bf16_t* Kb  = (bf16_t*)(ws + off); off += (size_t)BT * DK * 2;
    bf16_t* Vtb = (bf16_t*)(ws + off); off += (size_t)BATCH * DV * SEQ * 2;

    // 1) downconvert inputs to bf16
    {
        int n = BT * DMODEL;
        cvt_f32_bf16_kernel<<<(n + 255) / 256, 256, 0, stream>>>(x, xb, n);
        n = DK * DMODEL;
        cvt_f32_bf16_kernel<<<(n + 255) / 256, 256, 0, stream>>>(Wq, Wqb, n);
        cvt_f32_bf16_kernel<<<(n + 255) / 256, 256, 0, stream>>>(Wk, Wkb, n);
        cvt_f32_bf16_kernel<<<(n + 255) / 256, 256, 0, stream>>>(Wv, Wvb, n);
    }

    // 2) projections: (BT/32) row groups x (1024/64) col groups = 4096 waves
    {
        const int waves  = (BT / 32) * (1024 / 64);
        const int blocks = waves / 8;
        proj_gemm_kernel<<<blocks, 256, 0, stream>>>(xb, Wqb, Qb, 0);
        proj_gemm_kernel<<<blocks, 256, 0, stream>>>(xb, Wkb, Kb, 0);
        proj_gemm_kernel<<<blocks, 256, 0, stream>>>(xb, Wvb, Vtb, 1);
    }

    // 3) fused attention: one workgroup per 16 query rows per batch
    {
        const size_t smemBytes = (size_t)16 * SEQ * 4   // S (f32)
                               + (size_t)16 * SEQ * 2   // P (bf16)
                               + (256 + 16 + 16) * 4;   // reductions
        dim3 grid(SEQ / 16, BATCH);
        attn_kernel<<<grid, 256, smemBytes, stream>>>(Qb, Kb, Vtb, out);
    }
}
